// TextureEnhancedMDTA_29815662969217
// MI455X (gfx1250) — matrix operand order repaired and checked by reading, not compiled, not run on hardware
//
#include <hip/hip_runtime.h>
#include <cmath>

#define HW   16384
#define NB   8
#define CC   192
#define C3   576
#define HEADS 4
#define CH   48

typedef float  v2f  __attribute__((ext_vector_type(2)));
typedef float  v8f  __attribute__((ext_vector_type(8)));
typedef __bf16 v16bf __attribute__((ext_vector_type(16)));
typedef __bf16 vbf2  __attribute__((ext_vector_type(2)));

// Packed f32x2 -> bf16x2 (RNE); lowers to the native packed convert.
static __device__ __forceinline__ unsigned pack_bf16(float a, float b) {
    v2f f; f.x = a; f.y = b;
    vbf2 h = __builtin_convertvector(f, vbf2);
    return __builtin_bit_cast(unsigned, h);
}

// Fragment carrier: 8 packed-bf16 DWORDs == one v16bf WMMA operand.
union Frag16 {
    unsigned u[8];
    uint4    q[2];
    v16bf    v;
};

// ---------------------------------------------------------------------------
// 1x1 conv as GEMM: Y[b,m,p] = sum_k W[m,k] * X[b,k,p] + bias[m]
// Tile: 64(M) x 64(N) per 256-thread block; 8 waves, each 16x32 output.
// bf16 WMMA 16x16x32, f32 accumulate. LDS tiles hold packed bf16 pairs with
// 80B (16B-multiple) row stride so every fragment is 2x ds_load_b128.
// K is a compile-time constant so tile-fill addressing is immediate offsets.
// ---------------------------------------------------------------------------
template <int K>
__global__ __launch_bounds__(256) void gemm1x1_bf16wmma_kernel(
    const float* __restrict__ X, const float* __restrict__ W,
    const float* __restrict__ bias, float* __restrict__ Y, int M)
{
    __shared__ __align__(16) unsigned sW32[64][20];  // [m][kpair], kpair 0..15 used
    __shared__ __align__(16) unsigned sXT[64][20];   // [n][kpair] (transposed X tile)

    const int tid = threadIdx.x, lane = tid & 31, wave = tid >> 5;
    const int m0  = blockIdx.y * 64;
    const int nt  = blockIdx.x;
    const int bIdx = (nt * 64) / HW;
    const int p0   = (nt * 64) % HW;
    const float* Xb = X + (size_t)bIdx * K * HW;

    const int mw  = (wave & 3) * 16;
    const int nw  = (wave >> 2) * 32;
    const int hi  = lane >> 4;
    const int l15 = lane & 15;
    const int hi4 = hi * 4;      // unused (A fragment uses hi8u)
    const int hi8u = hi * 8;     // B-fragment uint offset within row

    // Per-thread tile-fill coordinates (constant strides -> immediate offsets)
    const int wmi = tid >> 4;    // W tile row group (0..15), rows wmi + 16*it
    const int wkp = tid & 15;    // W tile kpair
    const int xkp = tid >> 6;    // X tile kpair group (0..3), kpairs xkp + 4*it
    const int xni = tid & 63;    // X tile column
    const float* wbase = W + (size_t)(m0 + wmi) * K + 2 * wkp;
    const float* xbase = Xb + (size_t)(2 * xkp) * HW + p0 + xni;

    v8f acc0 = {}; v8f acc1 = {};

    for (int k0 = 0; k0 < K; k0 += 32) {
        #pragma unroll
        for (int it = 0; it < 4; ++it) {
            const float* wp = wbase + k0 + it * 16 * K;
            sW32[wmi + 16 * it][wkp] = pack_bf16(wp[0], wp[1]);
        }
        #pragma unroll
        for (int it = 0; it < 4; ++it) {
            const float* xp = xbase + (size_t)(k0 + 8 * it) * HW;
            sXT[xni][xkp + 4 * it] = pack_bf16(xp[0], xp[HW]);
        }
        if (wave == 0 && (k0 + 32) < K)   // global_prefetch next K rows
            __builtin_prefetch(Xb + (size_t)(k0 + 32 + lane) * HW + p0, 0, 0);
        __syncthreads();

        Frag16 a, b0, b1;
        // A 16x32 layout: per lane one 16-k run at k = 16*hi.
        const unsigned* wrow = &sW32[mw + l15][0];
        a.q[0] = *(const uint4*)(wrow + hi8u);
        a.q[1] = *(const uint4*)(wrow + hi8u + 4);
        // B 32x16 layout: per lane one 16-k run at k = 16*hi.
        const unsigned* x0 = &sXT[nw + l15][hi8u];
        b0.q[0] = *(const uint4*)(x0);
        b0.q[1] = *(const uint4*)(x0 + 4);
        const unsigned* x1 = &sXT[nw + 16 + l15][hi8u];
        b1.q[0] = *(const uint4*)(x1);
        b1.q[1] = *(const uint4*)(x1 + 4);

        acc0 = __builtin_amdgcn_wmma_f32_16x16x32_bf16(false, a.v, false, b0.v, (short)0, acc0, false, false);
        acc1 = __builtin_amdgcn_wmma_f32_16x16x32_bf16(false, a.v, false, b1.v, (short)0, acc1, false, false);
        __syncthreads();
    }

    #pragma unroll
    for (int r = 0; r < 8; ++r) {
        const int m = m0 + mw + r + 8 * hi;          // C/D layout: M = r + 8*hi
        const float bs = bias[m];
        const size_t base = ((size_t)bIdx * M + m) * HW + p0;
        Y[base + nw + l15]      = acc0[r] + bs;
        Y[base + nw + 16 + l15] = acc1[r] + bs;
    }
}

// ---------------------------------------------------------------------------
// Depthwise 3x3 (pad=1) + bias, fused with texture-boost scaling of V part.
// ---------------------------------------------------------------------------
__global__ __launch_bounds__(256) void dwconv3x3_kernel(
    const float* __restrict__ X,     // [B, 576, HW]
    const float* __restrict__ Wd,    // [576, 9]
    const float* __restrict__ bd,    // [576]
    const float* __restrict__ mask,  // [B, HW]
    const float* __restrict__ tboost,// scalar
    float* __restrict__ Y)           // [B, 576, HW]
{
    size_t idx = (size_t)blockIdx.x * 256 + threadIdx.x;
    if (idx >= (size_t)NB * C3 * HW) return;
    const int p = (int)(idx % HW);
    const size_t bc = idx / HW;
    const int ch = (int)(bc % C3);
    const int b  = (int)(bc / C3);
    const int hy = p >> 7, wx = p & 127;

    const float* src = X + bc * HW;
    const float* wt  = Wd + ch * 9;
    float s = 0.f;
    #pragma unroll
    for (int dy = -1; dy <= 1; ++dy) {
        const int yy = hy + dy;
        if (yy < 0 || yy >= 128) continue;
        #pragma unroll
        for (int dx = -1; dx <= 1; ++dx) {
            const int xx = wx + dx;
            if (xx < 0 || xx >= 128) continue;
            s += wt[(dy + 1) * 3 + (dx + 1)] * src[yy * 128 + xx];
        }
    }
    s += bd[ch];
    if (ch >= 2 * CC) {   // V channels: fuse (1 + mask*clip(tb,0.1,0.5))
        const float tb = fminf(fmaxf(tboost[0], 0.1f), 0.5f);
        s *= 1.f + mask[(size_t)b * HW + p] * tb;
    }
    Y[idx] = s;
}

// ---------------------------------------------------------------------------
// Per-batch texture mean -> per-(b,head) blended temperature.
// ---------------------------------------------------------------------------
__global__ __launch_bounds__(256) void texture_stats_kernel(
    const float* __restrict__ mask, const float* __restrict__ detail,
    const float* __restrict__ smooth, float* __restrict__ temps /*[B*HEADS]*/)
{
    const int b = blockIdx.x;
    __shared__ float red[256];
    float s = 0.f;
    for (int i = threadIdx.x; i < HW; i += 256) s += mask[(size_t)b * HW + i];
    red[threadIdx.x] = s; __syncthreads();
    for (int o = 128; o > 0; o >>= 1) {
        if (threadIdx.x < o) red[threadIdx.x] += red[threadIdx.x + o];
        __syncthreads();
    }
    if (threadIdx.x < HEADS) {
        const float tl = red[0] / (float)HW;
        temps[b * HEADS + threadIdx.x] =
            detail[threadIdx.x] * tl + smooth[threadIdx.x] * (1.f - tl);
    }
}

// ---------------------------------------------------------------------------
// Inverse L2 norms over spatial dim for q,k channels (384 per batch).
// ---------------------------------------------------------------------------
__global__ __launch_bounds__(256) void rownorm_kernel(
    const float* __restrict__ Y /*[B,576,HW]*/, float* __restrict__ rnorm /*[B,384]*/)
{
    const int ch = blockIdx.x;     // 0..383
    const int b  = blockIdx.y;
    const float* row = Y + ((size_t)b * C3 + ch) * HW;
    __shared__ float red[256];
    float s = 0.f;
    for (int i = threadIdx.x; i < HW; i += 256) { const float v = row[i]; s += v * v; }
    red[threadIdx.x] = s; __syncthreads();
    for (int o = 128; o > 0; o >>= 1) {
        if (threadIdx.x < o) red[threadIdx.x] += red[threadIdx.x + o];
        __syncthreads();
    }
    if (threadIdx.x == 0)
        rnorm[b * 2 * CC + ch] = 1.f / fmaxf(sqrtf(red[0]), 1e-12f);
}

// ---------------------------------------------------------------------------
// Attention per (b, head): S = (q_hat k_hat^T)*temp, softmax rows, O = S @ V.
// Phase A: bf16 WMMA on RAW q/k (row-norm scales applied as rank-1 scaling of
//          the accumulated Gram matrix), K split across 8 waves, reduced via
//          ds_add_f32 LDS atomics.
// Phase B: f32 WMMA 16x16x4, K=48; S fragment hoisted to registers.
// ---------------------------------------------------------------------------
__global__ __launch_bounds__(256) void attention_kernel(
    const float* __restrict__ QKV,   // [B,576,HW] (dw output, V pre-scaled)
    const float* __restrict__ rnorm, // [B,384]
    const float* __restrict__ temps, // [B,HEADS]
    float* __restrict__ O)           // [B,192,HW]
{
    const int b = blockIdx.x >> 2, h = blockIdx.x & 3;
    const int tid = threadIdx.x, lane = tid & 31, wave = tid >> 5;
    const int hi = lane >> 4, l15 = lane & 15, hi8 = hi * 8;

    const float* Q  = QKV + ((size_t)b * C3 + h * CH) * HW;
    const float* Kp = QKV + ((size_t)b * C3 + CC + h * CH) * HW;
    const float* V  = QKV + ((size_t)b * C3 + 2 * CC + h * CH) * HW;
    const float* rq = rnorm + b * 2 * CC + h * CH;
    const float* rk = rnorm + b * 2 * CC + CC + h * CH;

    __shared__ __align__(16) float sS[CH][CH + 2];   // 50-float rows: 8B-aligned pairs
    for (int i = tid; i < CH * (CH + 2); i += 256) (&sS[0][0])[i] = 0.f;
    __syncthreads();

    { // ---- Phase A: Gram matrix over this wave's K-slice (no in-loop scaling) ----
        v8f acc[3][3] = {};
        const float* qrow[3] = { Q + (size_t)l15 * HW, Q + (size_t)(16 + l15) * HW,
                                 Q + (size_t)(32 + l15) * HW };
        const float* krow[3] = { Kp + (size_t)l15 * HW, Kp + (size_t)(16 + l15) * HW,
                                 Kp + (size_t)(32 + l15) * HW };
        const int kwave = wave * 2048;
        for (int ks = 0; ks < 2048; ks += 32) {
            const int kb0 = kwave + ks;
            Frag16 af[3], bf3[3];
            #pragma unroll
            for (int t = 0; t < 3; ++t) {
                // A 16x32: two 8-k runs at kb0+hi8 and kb0+16+hi8 (b128 loads)
                const float4 qa = *(const float4*)(qrow[t] + kb0 + hi8);
                const float4 qb = *(const float4*)(qrow[t] + kb0 + hi8 + 4);
                const float4 qc = *(const float4*)(qrow[t] + kb0 + 16 + hi8);
                const float4 qd = *(const float4*)(qrow[t] + kb0 + 16 + hi8 + 4);
                af[t].u[0] = pack_bf16(qa.x, qa.y);
                af[t].u[1] = pack_bf16(qa.z, qa.w);
                af[t].u[2] = pack_bf16(qb.x, qb.y);
                af[t].u[3] = pack_bf16(qb.z, qb.w);
                af[t].u[4] = pack_bf16(qc.x, qc.y);
                af[t].u[5] = pack_bf16(qc.z, qc.w);
                af[t].u[6] = pack_bf16(qd.x, qd.y);
                af[t].u[7] = pack_bf16(qd.z, qd.w);
                // B 32x16: two 8-k runs at kb0+hi8 and kb0+16+hi8 (b128 loads)
                const float* kr = krow[t] + kb0 + hi8;
                const float4 ka = *(const float4*)(kr);
                const float4 kb = *(const float4*)(kr + 4);
                const float4 kc = *(const float4*)(kr + 16);
                const float4 kd = *(const float4*)(kr + 20);
                bf3[t].u[0] = pack_bf16(ka.x, ka.y);
                bf3[t].u[1] = pack_bf16(ka.z, ka.w);
                bf3[t].u[2] = pack_bf16(kb.x, kb.y);
                bf3[t].u[3] = pack_bf16(kb.z, kb.w);
                bf3[t].u[4] = pack_bf16(kc.x, kc.y);
                bf3[t].u[5] = pack_bf16(kc.z, kc.w);
                bf3[t].u[6] = pack_bf16(kd.x, kd.y);
                bf3[t].u[7] = pack_bf16(kd.z, kd.w);
            }
            #pragma unroll
            for (int mt = 0; mt < 3; ++mt)
                #pragma unroll
                for (int nt = 0; nt < 3; ++nt)
                    acc[mt][nt] = __builtin_amdgcn_wmma_f32_16x16x32_bf16(
                        false, af[mt].v, false, bf3[nt].v, (short)0, acc[mt][nt], false, false);
        }
        // Rank-1 row/column norm scaling applied once at the epilogue.
        const float rkc[3] = { rk[l15], rk[16 + l15], rk[32 + l15] };
        #pragma unroll
        for (int mt = 0; mt < 3; ++mt)
            #pragma unroll
            for (int r = 0; r < 8; ++r) {
                const int row = mt * 16 + r + 8 * hi;
                const float rqr = rq[row];
                #pragma unroll
                for (int nt = 0; nt < 3; ++nt)
                    atomicAdd(&sS[row][nt * 16 + l15], acc[mt][nt][r] * rqr * rkc[nt]);
            }
    }
    __syncthreads();

    // ---- temperature scale + row softmax (48 rows) ----
    if (tid < CH) {
        const float t = temps[b * HEADS + h];
        float mx = -3.0e38f;
        for (int j = 0; j < CH; ++j) { const float s = sS[tid][j] * t; sS[tid][j] = s; mx = fmaxf(mx, s); }
        float sum = 0.f;
        for (int j = 0; j < CH; ++j) { const float e = __expf(sS[tid][j] - mx); sS[tid][j] = e; sum += e; }
        const float inv = 1.f / sum;
        for (int j = 0; j < CH; ++j) sS[tid][j] *= inv;
    }
    __syncthreads();

    // ---- Phase B: O = S @ V using f32 WMMA 16x16x4 (K=48 = 12 steps) ----
    // Hoist the per-lane S fragment (72 floats) out of the n-tile loop.
    v2f aReg[3][12];
    #pragma unroll
    for (int kst = 0; kst < 12; ++kst) {
        const int kb = kst * 4 + hi * 2;             // 32-bit A: K split across halves
        #pragma unroll
        for (int mt = 0; mt < 3; ++mt)
            aReg[mt][kst] = *(const v2f*)&sS[mt * 16 + l15][kb];   // ds_load_b64
    }

    float* Ob = O + ((size_t)b * CC + h * CH) * HW;
    for (int ntile = wave; ntile < HW / 16; ntile += 8) {
        const int n0 = ntile * 16;
        const float* vcol = V + n0 + l15;
        v8f o0 = {}, o1 = {}, o2 = {};
        #pragma unroll
        for (int kst = 0; kst < 12; ++kst) {
            const int kb = kst * 4 + hi * 2;
            v2f bfr;
            bfr.x = vcol[(size_t)kb * HW];
            bfr.y = vcol[(size_t)(kb + 1) * HW];
            o0 = __builtin_amdgcn_wmma_f32_16x16x4_f32(false, aReg[0][kst], false, bfr, (short)0, o0, false, false);
            o1 = __builtin_amdgcn_wmma_f32_16x16x4_f32(false, aReg[1][kst], false, bfr, (short)0, o1, false, false);
            o2 = __builtin_amdgcn_wmma_f32_16x16x4_f32(false, aReg[2][kst], false, bfr, (short)0, o2, false, false);
        }
        #pragma unroll
        for (int r = 0; r < 8; ++r) {
            Ob[(size_t)(r + 8 * hi) * HW + n0 + l15]      = o0[r];
            Ob[(size_t)(16 + r + 8 * hi) * HW + n0 + l15] = o1[r];
            Ob[(size_t)(32 + r + 8 * hi) * HW + n0 + l15] = o2[r];
        }
    }
}

// ---------------------------------------------------------------------------
extern "C" void kernel_launch(void* const* d_in, const int* in_sizes, int n_in,
                              void* d_out, int out_size, void* d_ws, size_t ws_size,
                              hipStream_t stream)
{
    (void)in_sizes; (void)n_in; (void)out_size; (void)ws_size;
    const float* x      = (const float*)d_in[0];
    const float* mask   = (const float*)d_in[1];
    const float* qkv_w  = (const float*)d_in[2];
    const float* qkv_b  = (const float*)d_in[3];
    const float* dw_w   = (const float*)d_in[4];
    const float* dw_b   = (const float*)d_in[5];
    const float* out_w  = (const float*)d_in[6];
    const float* out_b  = (const float*)d_in[7];
    const float* dtemp  = (const float*)d_in[8];
    const float* stemp  = (const float*)d_in[9];
    const float* tboost = (const float*)d_in[10];
    float* out = (float*)d_out;

    char* ws = (char*)d_ws;
    const size_t SZ = (size_t)NB * C3 * HW * sizeof(float);   // 302 MB
    float* qkv_dw   = (float*)(ws);            // dw-conv output (q,k,v)
    float* qkv_lin  = (float*)(ws + SZ);       // 1x1 conv output (dead after dwconv)
    float* attn_out = qkv_lin;                 // reuse dead buffer
    float* temps    = (float*)(ws + 2 * SZ);           // [B*HEADS]
    float* rnorm    = (float*)(ws + 2 * SZ + 256);     // [B*384]

    // 1) qkv = conv1x1(x)            M=576, K=192, N=131072
    gemm1x1_bf16wmma_kernel<CC><<<dim3(2048, 9), 256, 0, stream>>>(x, qkv_w, qkv_b, qkv_lin, C3);
    // 2) depthwise 3x3 + bias (+ texture boost fused on V channels)
    dwconv3x3_kernel<<<dim3((NB * C3 * HW) / 256), 256, 0, stream>>>(qkv_lin, dw_w, dw_b, mask, tboost, qkv_dw);
    // 3) per-(b,head) temperature
    texture_stats_kernel<<<dim3(NB), 256, 0, stream>>>(mask, dtemp, stemp, temps);
    // 4) inverse L2 norms for q,k channels
    rownorm_kernel<<<dim3(2 * CC, NB), 256, 0, stream>>>(qkv_dw, rnorm);
    // 5) channel attention per (b, head)
    attention_kernel<<<dim3(NB * HEADS), 256, 0, stream>>>(qkv_dw, rnorm, temps, attn_out);
    // 6) final conv1x1               M=192, K=192, N=131072
    gemm1x1_bf16wmma_kernel<CC><<<dim3(2048, 3), 256, 0, stream>>>(attn_out, out_w, out_b, out, CC);
}
